// GlobalEncoderPp_24472723653373
// MI455X (gfx1250) — matrix operand order repaired and checked by reading, not compiled, not run on hardware
//
#include <hip/hip_runtime.h>
#include <hip/hip_bf16.h>

typedef __attribute__((ext_vector_type(16))) _Float16 v16h;
typedef __attribute__((ext_vector_type(8)))  float    v8f;
typedef _Float16 f16;

#define B_GRAPHS 4
#define N1 8192
#define S1 2048
#define S2 512
#define KNBR 128

// ---------- helpers -------------------------------------------------------

__device__ __forceinline__ float fast_tanh(float x) {
#if __has_builtin(__builtin_amdgcn_tanhf)
  return __builtin_amdgcn_tanhf(x);          // v_tanh_f32 (gfx1250 TRANS op)
#elif __has_builtin(__builtin_amdgcn_tanh_f32)
  return __builtin_amdgcn_tanh_f32(x);
#else
  return tanhf(x);
#endif
}

__device__ __forceinline__ unsigned fenc(float f) {
  unsigned u = __float_as_uint(f);
  return (u & 0x80000000u) ? ~u : (u | 0x80000000u);
}
__device__ __forceinline__ float fdec(unsigned u) {
  u = (u & 0x80000000u) ? (u & 0x7FFFFFFFu) : ~u;
  return __uint_as_float(u);
}

// A-operand 16x32 f16 tile from row-major LDS [rows][ldk] (ldk halves, 16B aligned rows).
// Lane L (0..15): M=L, K = {0..7, 16..23}; lanes 16..31: M=L-16, K = {8..15, 24..31}.
__device__ __forceinline__ v16h load_a16(const f16* base, int ldk, int k0, int lane) {
  const f16* p = base + (lane & 15) * ldk + k0 + ((lane & 16) ? 8 : 0);
  union { v16h v; uint4 q[2]; } u;
  u.q[0] = *(const uint4*)(p);
  u.q[1] = *(const uint4*)(p + 16);
  return u.v;
}

// B-operand fragment from packed weights: layout [kChunk][nTile][lane][16] halves.
// Two contiguous 16B loads per lane, 1KB contiguous per wave.
__device__ __forceinline__ v16h load_bp(const f16* __restrict__ Wp, int kc, int nt,
                                        int nTiles, int lane) {
  const f16* p = Wp + (((size_t)(kc * nTiles + nt) * 32) + lane) * 16;
  union { v16h v; uint4 q[2]; } u;
  u.q[0] = *(const uint4*)(p);
  u.q[1] = *(const uint4*)(p + 16);
  return u.v;
}

// ---------- weight conversion/packing: f32 [K][N] -> WMMA-tile f16 ---------
// Output element i: j=i&15, lane=(i>>4)&31, tile=i>>9, nt=tile%nTiles, kc=tile/nTiles
// holds W[kc*32 + kmap(lane,j)][nt*16 + (lane&15)] (0 beyond K).

__global__ void k_pack(const float* __restrict__ W, f16* __restrict__ O,
                       int K, int N, int nTiles, int tot) {
  int i = blockIdx.x * blockDim.x + threadIdx.x;
  if (i >= tot) return;
  int j = i & 15;
  int lane = (i >> 4) & 31;
  int tile = i >> 9;
  int nt = tile % nTiles;
  int kc = tile / nTiles;
  int n = nt * 16 + (lane & 15);
  int kk = ((lane & 16) ? 8 : 0) + ((j < 8) ? j : j + 8);
  int k = kc * 32 + kk;
  O[i] = (k < K) ? (f16)W[(size_t)k * N + n] : (f16)0.f;
}

// ---------- farthest point sampling (1 block per graph) --------------------

__global__ void __launch_bounds__(256) k_fps(const float* __restrict__ pos, int N, int S,
                                             int* __restrict__ samp,
                                             float* __restrict__ cent) {
  __shared__ float mind[N1];
  __shared__ float rv[256];
  __shared__ int   ri[256];
  __shared__ int   s_last;
  int b = blockIdx.x, tid = threadIdx.x;
  const float* P = pos + (size_t)b * N * 2;
  for (int i = tid; i < N; i += 256) mind[i] = 3.4e38f;
  if (tid == 0) s_last = 0;
  __syncthreads();
  for (int s = 0; s < S; ++s) {
    int last = s_last;
    float cx = P[last * 2], cy = P[last * 2 + 1];
    if (tid == 0) {
      samp[(size_t)b * S + s] = last;
      cent[((size_t)b * S + s) * 2 + 0] = cx;
      cent[((size_t)b * S + s) * 2 + 1] = cy;
    }
    float bv = -1.f; int bi = 0;
    for (int i = tid; i < N; i += 256) {
      float dx = P[i * 2] - cx, dy = P[i * 2 + 1] - cy;
      float d = dx * dx + dy * dy;
      float mn = fminf(mind[i], d);
      mind[i] = mn;
      if (mn > bv) { bv = mn; bi = i; }   // strict > keeps smallest index
    }
    rv[tid] = bv; ri[tid] = bi;
    __syncthreads();
    for (int off = 128; off > 0; off >>= 1) {
      if (tid < off) {
        float ov = rv[tid + off]; int oi = ri[tid + off];
        if (ov > rv[tid] || (ov == rv[tid] && oi < ri[tid])) { rv[tid] = ov; ri[tid] = oi; }
      }
      __syncthreads();
    }
    if (tid == 0) s_last = ri[0];
    __syncthreads();
  }
}

// ---------- fused set-abstraction: neighbors + PointNetConv + max ----------
// grid (S, B), 128 threads (4 waves). Dynamic LDS:
//   region0: d2[Npts] f32  (later reused as h1[128][F1] f16)
//   region1: A[128][KP] f16

template <int FIN, int KP, int F1, int F2>
__global__ void __launch_bounds__(128) k_sa(
    const float* __restrict__ feat, const float* __restrict__ apos,
    const float* __restrict__ cpos, float r2, int Npts,
    const f16* __restrict__ W1, const float* __restrict__ B1,
    const f16* __restrict__ W2, const float* __restrict__ B2,
    float* __restrict__ out) {
  constexpr int KC1 = KP / 32;   // k chunks, layer 1
  constexpr int NT1 = F1 / 16;   // n tiles,  layer 1
  constexpr int KC2 = F1 / 32;   // k chunks, layer 2
  constexpr int NT2 = F2 / 16;   // n tiles,  layer 2

  extern __shared__ char smem[];
  int S = gridDim.x;
  int s = blockIdx.x, b = blockIdx.y;
  int tid = threadIdx.x;
  int lane = tid & 31, wave = tid >> 5;

  float* d2 = (float*)smem;
  f16*   h1 = (f16*)smem;
  size_t r0 = (size_t)Npts * 4;
  size_t hb = (size_t)128 * F1 * 2;
  if (hb > r0) r0 = hb;
  f16* Amat = (f16*)(smem + r0);

  __shared__ int nidx[KNBR];
  __shared__ int scnt;
  __shared__ int ccnt;

  const float* P = apos + (size_t)b * Npts * 2;
  float cx = cpos[((size_t)b * S + s) * 2 + 0];
  float cy = cpos[((size_t)b * S + s) * 2 + 1];

  for (int i = tid; i < Npts; i += 128) {
    float dx = P[i * 2] - cx, dy = P[i * 2 + 1] - cy;
    d2[i] = dx * dx + dy * dy;
  }
  __syncthreads();

  auto count_le = [&](float t) -> int {
    if (tid == 0) scnt = 0;
    __syncthreads();
    int loc = 0;
    for (int i = tid; i < Npts; i += 128) loc += (d2[i] <= t) ? 1 : 0;
    atomicAdd(&scnt, loc);
    __syncthreads();
    int r = scnt;
    __syncthreads();
    return r;
  };

  // threshold = min(Kth-smallest d2, r2); exact via bit-pattern binary search
  float thr = r2;
  int c0 = count_le(r2);
  if (c0 > KNBR) {
    unsigned lo = 0, hi = __float_as_uint(r2);
    while (lo < hi) {
      unsigned mid = (lo + hi) >> 1;
      int cm = count_le(__uint_as_float(mid));
      if (cm >= KNBR) hi = mid; else lo = mid + 1;
    }
    thr = __uint_as_float(hi);
  }

  // compact neighbor indices (order-free: final max is permutation invariant)
  if (tid == 0) ccnt = 0;
  __syncthreads();
  for (int i = tid; i < Npts; i += 128) {
    if (d2[i] <= thr) {
      int p = atomicAdd(&ccnt, 1);
      if (p < KNBR) nidx[p] = i;
    }
  }
  __syncthreads();
  int m = ccnt; if (m > KNBR) m = KNBR;

  // build A[128][KP] = [x[nbr], rel, 0-pad] in f16 (one row per thread)
  {
    int row = tid;
    f16* ar = Amat + (size_t)row * KP;
    if (row < m) {
      int j = nidx[row];
      const float* fr = feat + ((size_t)b * Npts + j) * FIN;
#pragma unroll
      for (int f = 0; f < FIN; ++f) ar[f] = (f16)fr[f];
      ar[FIN + 0] = (f16)(P[j * 2 + 0] - cx);
      ar[FIN + 1] = (f16)(P[j * 2 + 1] - cy);
#pragma unroll
      for (int f = FIN + 2; f < KP; ++f) ar[f] = (f16)0.f;
    } else {
#pragma unroll
      for (int f = 0; f < KP; ++f) ar[f] = (f16)0.f;
    }
  }
  __syncthreads();

  // layer 1: h1 = tanh(A @ W1 + b1)   [128, F1]  (h1 overlays dead d2)
  for (int nt = wave; nt < NT1; nt += 4) {
    int coln = nt * 16 + (lane & 15);
    float bias = B1[coln];
    v16h bt[KC1];
#pragma unroll
    for (int kc = 0; kc < KC1; ++kc) bt[kc] = load_bp(W1, kc, nt, NT1, lane);
#pragma unroll
    for (int mt = 0; mt < 8; ++mt) {
      v8f acc = {};
#pragma unroll
      for (int kc = 0; kc < KC1; ++kc) {
        v16h a = load_a16(Amat + (size_t)(mt * 16) * KP, KP, kc * 32, lane);
        acc = __builtin_amdgcn_wmma_f32_16x16x32_f16(false, a, false, bt[kc],
                                                     (short)0, acc, false, false);
      }
      int rbase = mt * 16 + ((lane & 16) ? 8 : 0);
#pragma unroll
      for (int r = 0; r < 8; ++r)
        h1[(size_t)(rbase + r) * F1 + coln] = (f16)fast_tanh(acc[r] + bias);
    }
  }
  __syncthreads();

  // layer 2 + masked max over neighbor rows
  for (int nt = wave; nt < NT2; nt += 4) {
    int coln = nt * 16 + (lane & 15);
    float cmax = -3.4e38f;
    v16h bt[KC2];
#pragma unroll
    for (int kc = 0; kc < KC2; ++kc) bt[kc] = load_bp(W2, kc, nt, NT2, lane);
#pragma unroll
    for (int mt = 0; mt < 8; ++mt) {
      v8f acc = {};
#pragma unroll
      for (int kc = 0; kc < KC2; ++kc) {
        v16h a = load_a16(h1 + (size_t)(mt * 16) * F1, F1, kc * 32, lane);
        acc = __builtin_amdgcn_wmma_f32_16x16x32_f16(false, a, false, bt[kc],
                                                     (short)0, acc, false, false);
      }
      int rbase = mt * 16 + ((lane & 16) ? 8 : 0);
#pragma unroll
      for (int r = 0; r < 8; ++r)
        if (rbase + r < m) cmax = fmaxf(cmax, acc[r]);
    }
    cmax = fmaxf(cmax, __shfl_xor(cmax, 16, 32));
    if (lane < 16)
      out[((size_t)b * S + s) * F2 + coln] = cmax + B2[coln];
  }
}

// ---------- global MLP + global max pool (1 block per graph) ---------------
// 128 threads (4 waves); per-wave A[16][288] and H[16][256] tiles in LDS.

__global__ void __launch_bounds__(128) k_glob(
    const float* __restrict__ x2, const float* __restrict__ p2,
    const f16* __restrict__ W1, const float* __restrict__ B1,
    const f16* __restrict__ W2, const float* __restrict__ B2,
    float* __restrict__ out) {
  extern __shared__ char smem[];
  int b = blockIdx.x, tid = threadIdx.x;
  int lane = tid & 31, wave = tid >> 5;
  f16* Aall = (f16*)smem;
  f16* Hall = (f16*)(smem + (size_t)4 * 16 * 288 * 2);
  f16* Aw = Aall + (size_t)wave * 16 * 288;
  f16* Hw = Hall + (size_t)wave * 16 * 256;
  __shared__ unsigned colmax[512];
  for (int i = tid; i < 512; i += 128) colmax[i] = fenc(-3.4e38f);
  __syncthreads();

  const float* X = x2 + (size_t)b * S2 * 256;
  const float* P = p2 + (size_t)b * S2 * 2;

  for (int mt = wave; mt < 32; mt += 4) {   // 32 row tiles of 16 (512 rows)
    // build A tile: [x2 row (256), pos2 (2), 0-pad to 288]
    for (int idx = lane; idx < 16 * 288; idx += 32) {
      int rr = idx / 288, k = idx - rr * 288;
      int row = mt * 16 + rr;
      float v = 0.f;
      if (k < 256)      v = X[(size_t)row * 256 + k];
      else if (k < 258) v = P[row * 2 + (k - 256)];
      Aw[idx] = (f16)v;
    }
    __syncthreads();
    // layer 1: [16,288] x [288,256] -> tanh -> Hw
    for (int nt = 0; nt < 16; ++nt) {
      int coln = nt * 16 + (lane & 15);
      v8f acc = {};
#pragma unroll
      for (int kc = 0; kc < 9; ++kc) {
        v16h a  = load_a16(Aw, 288, kc * 32, lane);
        v16h bm = load_bp(W1, kc, nt, 16, lane);
        acc = __builtin_amdgcn_wmma_f32_16x16x32_f16(false, a, false, bm,
                                                     (short)0, acc, false, false);
      }
      float bias = B1[coln];
      int rbase = (lane & 16) ? 8 : 0;
#pragma unroll
      for (int r = 0; r < 8; ++r)
        Hw[(size_t)(rbase + r) * 256 + coln] = (f16)fast_tanh(acc[r] + bias);
    }
    __syncthreads();
    // layer 2: [16,256] x [256,512] -> column max (monotone-uint atomic max)
    for (int nt = 0; nt < 32; ++nt) {
      int coln = nt * 16 + (lane & 15);
      v8f acc = {};
#pragma unroll
      for (int kc = 0; kc < 8; ++kc) {
        v16h a  = load_a16(Hw, 256, kc * 32, lane);
        v16h bm = load_bp(W2, kc, nt, 32, lane);
        acc = __builtin_amdgcn_wmma_f32_16x16x32_f16(false, a, false, bm,
                                                     (short)0, acc, false, false);
      }
      float lm = acc[0];
#pragma unroll
      for (int r = 1; r < 8; ++r) lm = fmaxf(lm, acc[r]);
      lm = fmaxf(lm, __shfl_xor(lm, 16, 32));
      if (lane < 16) atomicMax(&colmax[coln], fenc(lm + B2[coln]));
    }
    __syncthreads();
  }
  for (int i = tid; i < 512; i += 128)
    out[(size_t)b * 512 + i] = fdec(colmax[i]);
}

// ---------- host launch ----------------------------------------------------

static inline size_t alignup(size_t x) { return (x + 255) & ~(size_t)255; }

extern "C" void kernel_launch(void* const* d_in, const int* in_sizes, int n_in,
                              void* d_out, int out_size, void* d_ws, size_t ws_size,
                              hipStream_t stream) {
  const float* x     = (const float*)d_in[0];   // [4,8192,8]
  const float* pos   = (const float*)d_in[1];   // [4,8192,2]
  const float* l1_w1 = (const float*)d_in[2];
  const float* l1_b1 = (const float*)d_in[3];
  const float* l1_w2 = (const float*)d_in[4];
  const float* l1_b2 = (const float*)d_in[5];
  const float* l2_w1 = (const float*)d_in[6];
  const float* l2_b1 = (const float*)d_in[7];
  const float* l2_w2 = (const float*)d_in[8];
  const float* l2_b2 = (const float*)d_in[9];
  const float* g_w1  = (const float*)d_in[10];
  const float* g_b1  = (const float*)d_in[11];
  const float* g_w2  = (const float*)d_in[12];
  const float* g_b2  = (const float*)d_in[13];
  float* out = (float*)d_out;                   // [4,1,512]

  // workspace carve
  char* ws = (char*)d_ws;
  size_t off = 0;
  auto take = [&](size_t bytes) { char* p = ws + off; off = alignup(off + bytes); return p; };
  f16* w1a = (f16*)take((size_t)32 * 64 * 2);     // 10->32 pad, 64
  f16* w1b = (f16*)take((size_t)64 * 128 * 2);
  f16* w2a = (f16*)take((size_t)160 * 128 * 2);   // 130->160 pad
  f16* w2b = (f16*)take((size_t)128 * 256 * 2);
  f16* wga = (f16*)take((size_t)288 * 256 * 2);   // 258->288 pad
  f16* wgb = (f16*)take((size_t)256 * 512 * 2);
  int*   samp1 = (int*)take((size_t)B_GRAPHS * S1 * 4);
  float* cent1 = (float*)take((size_t)B_GRAPHS * S1 * 2 * 4);
  float* x1    = (float*)take((size_t)B_GRAPHS * S1 * 128 * 4);
  int*   samp2 = (int*)take((size_t)B_GRAPHS * S2 * 4);
  float* cent2 = (float*)take((size_t)B_GRAPHS * S2 * 2 * 4);
  float* x2    = (float*)take((size_t)B_GRAPHS * S2 * 256 * 4);
  (void)ws_size; (void)in_sizes; (void)n_in; (void)out_size;

  // allow >64KB dynamic LDS for the fused kernels
  hipFuncSetAttribute((const void*)(k_sa<8, 32, 64, 128>),
                      hipFuncAttributeMaxDynamicSharedMemorySize, 96 * 1024);
  hipFuncSetAttribute((const void*)(k_sa<128, 160, 128, 256>),
                      hipFuncAttributeMaxDynamicSharedMemorySize, 96 * 1024);
  hipFuncSetAttribute((const void*)k_glob,
                      hipFuncAttributeMaxDynamicSharedMemorySize, 96 * 1024);

  // 1) convert + pack weights into WMMA tile layout (f16, K padded to 32)
  auto pack = [&](const float* W, f16* O, int K, int Kpad, int N) {
    int tot = Kpad * N;  // == (Kpad/32)*(N/16)*512
    k_pack<<<(tot + 255) / 256, 256, 0, stream>>>(W, O, K, N, N / 16, tot);
  };
  pack(l1_w1, w1a, 10, 32, 64);
  pack(l1_w2, w1b, 64, 64, 128);
  pack(l2_w1, w2a, 130, 160, 128);
  pack(l2_w2, w2b, 128, 128, 256);
  pack(g_w1,  wga, 258, 288, 256);
  pack(g_w2,  wgb, 256, 256, 512);

  // 2) level-1 FPS + fused set abstraction
  k_fps<<<B_GRAPHS, 256, 0, stream>>>(pos, N1, S1, samp1, cent1);
  {
    size_t dyn = (size_t)N1 * 4 + (size_t)128 * 32 * 2;   // d2/h1 + A
    dim3 grid(S1, B_GRAPHS);
    k_sa<8, 32, 64, 128><<<grid, 128, dyn, stream>>>(
        x, pos, cent1, 0.3f * 0.3f, N1, w1a, l1_b1, w1b, l1_b2, x1);
  }

  // 3) level-2 FPS + fused set abstraction (candidates = level-1 centers)
  k_fps<<<B_GRAPHS, 256, 0, stream>>>(cent1, S1, S2, samp2, cent2);
  {
    size_t dyn = (size_t)128 * 128 * 2 + (size_t)128 * 160 * 2;  // h1(>=d2) + A
    dim3 grid(S2, B_GRAPHS);
    k_sa<128, 160, 128, 256><<<grid, 128, dyn, stream>>>(
        x1, cent1, cent2, 0.6f * 0.6f, S1, w2a, l2_b1, w2b, l2_b2, x2);
  }

  // 4) global MLP + global max pool
  {
    size_t dyn = (size_t)4 * 16 * 288 * 2 + (size_t)4 * 16 * 256 * 2;
    k_glob<<<B_GRAPHS, 128, dyn, stream>>>(x2, cent2, wga, g_b1, wgb, g_b2, out);
  }
}
